// LocalWrapper_33397665693796
// MI455X (gfx1250) — matrix operand order repaired and checked by reading, not compiled, not run on hardware
//
#include <hip/hip_runtime.h>
#include <float.h>

typedef __attribute__((ext_vector_type(16))) _Float16 v16h;
typedef __attribute__((ext_vector_type(8)))  _Float16 v8h;
typedef __attribute__((ext_vector_type(4)))  _Float16 v4h;
typedef __attribute__((ext_vector_type(8)))  float    v8f;

#define N_TOK 50176   // 16 * 56 * 56
#define CDIM  512
#define BM 128
#define BN 128
#define BK 32
#define LDT 40        // padded LDS tile stride in halves (32 + 8)
#define WSZ 7
#define LK  516       // padded LDS row stride (floats) for attention

// ---------------------------------------------------------------------------
// Fused QKV projection GEMM: out = x @ w + bias, fp32 in, fp16 WMMA, fp32 out.
// blockIdx.z in {0,1,2} selects (wq,bq,q) / (wk,bk,k) / (wv,bv,v).
// Block tile 128x128, K-step 32. 8 waves (wave32) in a 2(M) x 4(N) grid,
// each wave computes a 64x32 tile via 4x2 v_wmma_f32_16x16x32_f16 accums.
//
// Fragment layouts (CDNA5 ISA 7.12.2):
//   A 16x32 f16: lanes 0-15 -> M=lane, K 0-7 in v0-3, K 16-23 in v4-7;
//                lanes 16-31 -> K 8-15, K 24-31.           (interleaved runs)
//   B 32x16 f16: lanes 0-15 -> N=lane, K 0-7, K 16-23;
//                lanes 16-31 -> K 8-15, K 24-31.           (interleaved runs)
//   C/D 16x16 f32: VGPR r -> M = (lane>=16 ? 8 : 0)+r, N = lane&15.
// ---------------------------------------------------------------------------
__global__ __launch_bounds__(256) void qkv_gemm(
    const float* __restrict__ x,
    const float* __restrict__ wq, const float* __restrict__ bq,
    const float* __restrict__ wk, const float* __restrict__ bk,
    const float* __restrict__ wv, const float* __restrict__ bv,
    float* __restrict__ qf, float* __restrict__ kf, float* __restrict__ vf)
{
    const float* w;
    const float* bias;
    float* out;
    if (blockIdx.z == 0)      { w = wq; bias = bq; out = qf; }
    else if (blockIdx.z == 1) { w = wk; bias = bk; out = kf; }
    else                      { w = wv; bias = bv; out = vf; }

    __shared__ _Float16 As[BM * LDT];   // As[m][k], row-major, padded
    __shared__ _Float16 Bs[BN * LDT];   // Bs[n][k], i.e. B transposed, padded

    const int tid   = threadIdx.x;
    const int lane  = tid & 31;
    const int wid   = tid >> 5;
    const int waveM = wid & 1;          // 0..1  -> 64-row slab
    const int waveN = wid >> 1;         // 0..3  -> 32-col slab
    const int m0    = blockIdx.x * BM;
    const int n0    = blockIdx.y * BN;

    v8f acc[4][2];
#pragma unroll
    for (int fm = 0; fm < 4; ++fm)
#pragma unroll
        for (int fn = 0; fn < 2; ++fn)
#pragma unroll
            for (int u = 0; u < 8; ++u) acc[fm][fn][u] = 0.0f;

    const int kbA  = (lane >> 4) * 8;   // A: interleaved 8-half runs at kbA, kbA+16
    const int kbB  = (lane >> 4) * 8;   // B: interleaved 8-half runs at kbB, kbB+16
    const int arow = waveM * 64 + (lane & 15);
    const int bcol = waveN * 32 + (lane & 15);

    for (int kk = 0; kk < CDIM; kk += BK) {
        // --- stage A tile (128x32 fp32 -> fp16), 1024 float4 / 256 threads ---
#pragma unroll
        for (int i = 0; i < 4; ++i) {
            int idx = tid + i * 256;        // float4 index
            int row = idx >> 3;             // 8 float4 per 32-float row
            int c4  = idx & 7;
            const float4 f = *(const float4*)(x + (size_t)(m0 + row) * CDIM + kk + c4 * 4);
            v4h h;
            h[0] = (_Float16)f.x; h[1] = (_Float16)f.y;
            h[2] = (_Float16)f.z; h[3] = (_Float16)f.w;
            *(v4h*)(As + row * LDT + c4 * 4) = h;
        }
        // --- stage B tile transposed (w[k][n] -> Bs[n][k]) ---
#pragma unroll
        for (int i = 0; i < 4; ++i) {
            int idx = tid + i * 256;        // 32 k-rows x 32 float4
            int kr  = idx >> 5;
            int c4  = idx & 31;
            const float4 f = *(const float4*)(w + (size_t)(kk + kr) * CDIM + n0 + c4 * 4);
            Bs[(c4 * 4 + 0) * LDT + kr] = (_Float16)f.x;
            Bs[(c4 * 4 + 1) * LDT + kr] = (_Float16)f.y;
            Bs[(c4 * 4 + 2) * LDT + kr] = (_Float16)f.z;
            Bs[(c4 * 4 + 3) * LDT + kr] = (_Float16)f.w;
        }
        __syncthreads();

        v16h afr[4];
#pragma unroll
        for (int fm = 0; fm < 4; ++fm) {
            const _Float16* p = As + (arow + fm * 16) * LDT;
            v8h lo = *(const v8h*)(p + kbA);
            v8h hi = *(const v8h*)(p + kbA + 16);
            afr[fm] = __builtin_shufflevector(lo, hi, 0,1,2,3,4,5,6,7,8,9,10,11,12,13,14,15);
        }
        v16h bfr[2];
#pragma unroll
        for (int fn = 0; fn < 2; ++fn) {
            const _Float16* p = Bs + (bcol + fn * 16) * LDT + kbB;
            v8h lo = *(const v8h*)(p);
            v8h hi = *(const v8h*)(p + 16);
            bfr[fn] = __builtin_shufflevector(lo, hi, 0,1,2,3,4,5,6,7,8,9,10,11,12,13,14,15);
        }
#pragma unroll
        for (int fm = 0; fm < 4; ++fm)
#pragma unroll
            for (int fn = 0; fn < 2; ++fn)
                acc[fm][fn] = __builtin_amdgcn_wmma_f32_16x16x32_f16(
                    false, afr[fm], false, bfr[fn],
                    (short)0, acc[fm][fn], false, false);
        __syncthreads();
    }

    // Epilogue: C/D layout: VGPR r -> (M = base + r), N = lane&15.
    const int mbase = m0 + waveM * 64 + (lane >> 4) * 8;
    const int nbase = n0 + waveN * 32 + (lane & 15);
#pragma unroll
    for (int fm = 0; fm < 4; ++fm)
#pragma unroll
        for (int fn = 0; fn < 2; ++fn) {
            const int n  = nbase + fn * 16;
            const float bb = bias[n];
#pragma unroll
            for (int r = 0; r < 8; ++r) {
                const int m = mbase + fm * 16 + r;
                out[(size_t)m * CDIM + n] = acc[fm][fn][r] + bb;
            }
        }
}

// ---------------------------------------------------------------------------
// Local attention, one block per (window, batch). window=7, look-around -> 14
// keys per query. Shared-qk: k rows L2-normalized (clip 1e-12); self score
// -5e4; causal + pad masks -FLT_MAX; softmax; p @ v.
// ---------------------------------------------------------------------------
__global__ __launch_bounds__(256) void local_attn(
    const float* __restrict__ qf, const float* __restrict__ kf,
    const float* __restrict__ vf, float* __restrict__ outp)
{
    __shared__ float sq[WSZ * LK];
    __shared__ float sk[14 * LK];
    __shared__ float spart[14 * 16];
    __shared__ float sscale[14];
    __shared__ float sdots[WSZ * 16];
    __shared__ float sp[WSZ * 16];

    const int b   = blockIdx.y;
    const int wi  = blockIdx.x;
    const int t0  = wi * WSZ;
    const int tid = threadIdx.x;
    const size_t base = (size_t)b * 3136 * CDIM;

    // stage q rows
    for (int idx = tid; idx < WSZ * CDIM; idx += 256) {
        int i = idx >> 9, e = idx & (CDIM - 1);
        sq[i * LK + e] = qf[base + (size_t)(t0 + i) * CDIM + e];
    }
    // stage k rows (zero-fill the padded / out-of-range rows)
    for (int idx = tid; idx < 14 * CDIM; idx += 256) {
        int j = idx >> 9, e = idx & (CDIM - 1);
        int kg = t0 - WSZ + j;
        sk[j * LK + e] = (kg >= 0) ? kf[base + (size_t)kg * CDIM + e] : 0.0f;
    }
    __syncthreads();

    // k row norms: 14 rows x 16 segments of 32 elems
    if (tid < 224) {
        int j = tid >> 4, seg = tid & 15;
        const float* p = sk + j * LK + seg * 32;
        float s = 0.0f;
#pragma unroll
        for (int u = 0; u < 32; ++u) s += p[u] * p[u];
        spart[j * 16 + seg] = s;
    }
    __syncthreads();
    if (tid < 14) {
        float s = 0.0f;
#pragma unroll
        for (int u = 0; u < 16; ++u) s += spart[tid * 16 + u];
        sscale[tid] = 1.0f / fmaxf(sqrtf(s), 1e-12f);
    }
    __syncthreads();

    // 7x14 score matrix, one (i,j) pair per thread
    if (tid < WSZ * 14) {
        int i = tid / 14, j = tid % 14;
        const float* pq = sq + i * LK;
        const float* pk = sk + j * LK;
        float d = 0.0f;
        for (int e = 0; e < CDIM; ++e) d += pq[e] * pk[e];
        int qg = t0 + i, kg = t0 - WSZ + j;
        float s;
        if (kg < 0 || kg > qg) s = -FLT_MAX;                             // pad / causal
        else if (kg == qg)     s = -5e4f;                                // shared-qk self
        else                   s = d * sscale[j] * 0.04419417382415922f; // * 512^-0.5
        sdots[i * 16 + j] = s;
    }
    __syncthreads();

    // softmax per query row
    if (tid < WSZ) {
        float m = -FLT_MAX;
#pragma unroll
        for (int j = 0; j < 14; ++j) m = fmaxf(m, sdots[tid * 16 + j]);
        float pv[14];
        float sum = 0.0f;
#pragma unroll
        for (int j = 0; j < 14; ++j) {
            float e = __expf(sdots[tid * 16 + j] - m);
            pv[j] = e;
            sum += e;
        }
        float inv = 1.0f / sum;
#pragma unroll
        for (int j = 0; j < 14; ++j) sp[tid * 16 + j] = pv[j] * inv;
    }
    __syncthreads();

    // out = p @ v  (v streamed from global, coalesced, hot in L2)
    for (int idx = tid; idx < WSZ * CDIM; idx += 256) {
        int i = idx >> 9, e = idx & (CDIM - 1);
        float a = 0.0f;
#pragma unroll
        for (int j = 0; j < 14; ++j) {
            int kg = t0 - WSZ + j;
            if (kg >= 0) a += sp[i * 16 + j] * vf[base + (size_t)kg * CDIM + e];
        }
        outp[base + (size_t)(t0 + i) * CDIM + e] = a;
    }
}

extern "C" void kernel_launch(void* const* d_in, const int* in_sizes, int n_in,
                              void* d_out, int out_size, void* d_ws, size_t ws_size,
                              hipStream_t stream) {
    const float* x  = (const float*)d_in[0];
    const float* wq = (const float*)d_in[1];
    const float* bq = (const float*)d_in[2];
    const float* wk = (const float*)d_in[3];
    const float* bk = (const float*)d_in[4];
    const float* wv = (const float*)d_in[5];
    const float* bv = (const float*)d_in[6];
    float* out = (float*)d_out;

    const size_t S = (size_t)N_TOK * CDIM;
    float* qf = (float*)d_ws;
    float* kf = qf + S;
    float* vf = kf + S;

    dim3 gg(N_TOK / BM, CDIM / BN, 3);     // 392 x 4 x 3
    qkv_gemm<<<gg, 256, 0, stream>>>(x, wq, bq, wk, bk, wv, bv, qf, kf, vf);

    dim3 ga(3136 / WSZ, 16);               // 448 x 16
    local_attn<<<ga, 256, 0, stream>>>(qf, kf, vf, out);
}
